// ScannedRNN_73985106641015
// MI455X (gfx1250) — matrix operand order repaired and checked by reading, not compiled, not run on hardware
//
#include <hip/hip_runtime.h>

// ---------------------------------------------------------------------------
// GRU scan for MI455X (gfx1250, wave32, WMMA), persistent-workgroup design.
//
//   r = sigmoid(x@Wir + bir + h@Whr)        -> fused K=512 GEMM [x;h]@[Wi;Wh]
//   z = sigmoid(x@Wiz + biz + h@Whz)        -> fused K=512 GEMM
//   n = tanh  (x@Win + bin + r*(h@Whn+bhn)) -> two K=256 GEMMs (cannot fuse)
//   h' = (1-z)*n + z*h ;  reset zeroes h before the cell.
//
// Virtual output dim N' = 1024 columns: [ r(256) | z(256) | i_n(256) | h_n(256) ]
//   n' <  512 : K 0..511 (x and h) | [512,768): K 0..255 (x) | >=768: K 256..511 (h)
//
// 8 workgroups x 512 threads (16 waves); each WG owns 32 batch rows and loops
// over all T=512 steps (recurrence independent across batch -> no grid sync).
// Waves 0-7: r/z tiles (full K). Waves 8-11: i_n tiles. Waves 12-15: h_n tiles.
// Per wave: 8 output tiles, kc-outer / tile-inner with 8 independent WMMA accs.
// ---------------------------------------------------------------------------

typedef _Float16 f16;
typedef __attribute__((ext_vector_type(4)))  _Float16 v4h;
typedef __attribute__((ext_vector_type(16))) _Float16 v16h;
typedef __attribute__((ext_vector_type(4)))  float    v4f;
typedef __attribute__((ext_vector_type(8)))  float    v8f;

#define T_STEPS 512
#define BB      256
#define HH      256
#define KK      512                // fused K ([x ; h])
#define NP      1024               // virtual N' (r|z|in|hn)
#define MT      32                 // batch rows per workgroup
#define NWG     (BB / MT)          // 8 workgroups
#define THREADS 512                // 16 wave32
#define NJ      (NP / 16)          // 64 N'-tiles
#define KCN     (KK / 32)          // 16 K-chunks of 32

// LDS: A = [2 m-tiles][16 kc][32 lanes][16 halves] f16 (32 KB, WMMA-swizzled)
//      G = [32 rows][GSTRIDE] f32 gate pre-activations (128.5 KB, padded)
#define A_HALVES (2 * KCN * 32 * 16)
#define GSTRIDE  1028
#define LDS_BYTES (A_HALVES * 2 + MT * GSTRIDE * 4)

// ---- WMMA fragment index math (ISA 7.12.2, 16-bit matrices, wave32) --------
// A 16x32: lane<16: halves 0..7 -> K 0..7, 8..15 -> K 16..23; lane>=16: +8.
__device__ __forceinline__ int a_off(int m, int k) {
  int mt = m >> 4, mr = m & 15;
  int kc = k >> 5, k5 = k & 31;
  int hi = (k5 >> 3) & 1;
  int e  = (k5 & 7) | ((k5 & 16) >> 1);
  return (((mt * KCN + kc) * 32) + (mr | (hi << 4))) * 16 + e;
}
// B 32x16: lane<16: halves 0..7 -> K 0..7, 8..15 -> K 16..23; lane>=16: +8; n = lane&15.
__device__ __forceinline__ int b_k(int lane, int e) {
  return ((e & 8) << 1) + (((lane >> 4) & 1) << 3) + (e & 7);
}

// ---------------------------------------------------------------------------
// Kernel 1: build swizzled f16 weight buffer in d_ws (1 MB, L2-resident).
// ---------------------------------------------------------------------------
__global__ void build_wt(const float* __restrict__ Wi,
                         const float* __restrict__ Wh,
                         f16* __restrict__ wt) {
  int idx  = blockIdx.x * blockDim.x + threadIdx.x;   // < 524288
  int e    = idx & 15;
  int lane = (idx >> 4) & 31;
  int kc   = (idx >> 9) & (KCN - 1);
  int j    = idx >> 13;
  int np_  = j * 16 + (lane & 15);
  int k    = kc * 32 + b_k(lane, e);
  float v;
  if (np_ < 512)      v = (k < 256) ? Wi[k * 768 + np_] : Wh[(k - 256) * 768 + np_];
  else if (np_ < 768) v = (k < 256) ? Wi[k * 768 + np_] : 0.0f;
  else                v = (k >= 256) ? Wh[(k - 256) * 768 + (np_ - 256)] : 0.0f;
  wt[idx] = (f16)v;
}

// ---------------------------------------------------------------------------
// Kernel 2: persistent GRU scan.
// ---------------------------------------------------------------------------
__global__ __launch_bounds__(THREADS, 1)
void gru_scan(const float* __restrict__ ins,     // [T,B,H] f32
              const int*   __restrict__ resets,  // [T,B]
              const float* __restrict__ carry,   // [B,H]
              const f16*   __restrict__ wt,      // swizzled [NJ][KCN][32][16]
              const float* __restrict__ bi,      // [3H]
              const float* __restrict__ bhn,     // [H]
              float*       __restrict__ out)     // [T,B,H]
{
  extern __shared__ __align__(32) char smem[];
  f16*   As = (f16*)smem;                        // swizzled A = [x_t ; h_eff]
  float* Gs = (float*)(smem + A_HALVES * 2);     // gate pre-activations

  const int tid  = threadIdx.x;
  const int lane = tid & 31;
  const int w    = tid >> 5;
  const int wu   = __builtin_amdgcn_readfirstlane(w);  // scalar wave id
  const int mt_w = w & 1;
  const int j0   = (w >> 1) * 8;
  const int base = blockIdx.x * MT;
  // Wave-uniform, SGPR-resident K range -> scalar loop, EXEC stays all-ones.
  const int kc0 = (wu >= 12) ? (KCN / 2) : 0;
  const int kc1 = (wu >= 8 && wu < 12) ? (KCN / 2) : KCN;

  // ---- prologue: stage x_0 and reset-masked carry into A (swizzled f16) ----
  {
    const float* x0 = ins + (size_t)base * HH;
#pragma unroll
    for (int i = 0; i < 4; ++i) {
      int flat = (tid + i * THREADS) * 4;          // 4 consecutive elements
      int m = flat >> 8, n = flat & 255;
      int rst = resets[base + m];
      v4f cv = *(const v4f*)(carry + (base + m) * HH + n);
      v4f xv = *(const v4f*)(x0 + flat);
      v4h hp, xp;
#pragma unroll
      for (int q = 0; q < 4; ++q) { hp[q] = rst ? (f16)0.0f : (f16)cv[q]; xp[q] = (f16)xv[q]; }
      *(v4h*)(As + a_off(m, 256 + n)) = hp;
      *(v4h*)(As + a_off(m, n))       = xp;
    }
  }
  __syncthreads();

  for (int t = 0; t < T_STEPS; ++t) {
    // ------------- Phase B: WMMA GEMM into G (kc outer, 8 indep accs) -------
    v8f acc[8] = {};
    {
      const f16* bbase = wt + (size_t)j0 * (KCN * 512) + lane * 16;
      for (int kc = kc0; kc < kc1; ++kc) {
        const v16h a = *(const v16h*)(As + ((mt_w * KCN + kc) * 32 + lane) * 16);
        const f16* bp = bbase + kc * 512;
#pragma unroll
        for (int jj = 0; jj < 8; ++jj) {
          const v16h b = *(const v16h*)(bp + (size_t)jj * (KCN * 512));
          acc[jj] = __builtin_amdgcn_wmma_f32_16x16x32_f16(
              false, a, false, b, (short)0, acc[jj], false, false);
        }
      }
    }
    {
      const int nloc = lane & 15;
      const int m0   = mt_w * 16 + ((lane >= 16) ? 8 : 0);
#pragma unroll
      for (int jj = 0; jj < 8; ++jj) {
        int   np_  = (j0 + jj) * 16 + nloc;
        float bias = (np_ < 768) ? bi[np_] : bhn[np_ - 768];
#pragma unroll
        for (int v = 0; v < 8; ++v)
          Gs[(m0 + v) * GSTRIDE + np_] = acc[jj][v] + bias;
      }
    }
    __syncthreads();

    // ------------- Phase C: gates + h' + output, stage x/h for t+1 ---------
    float* outt = out + ((size_t)t * BB + base) * HH;
    const float* xn = ins + ((size_t)(t + 1) * BB + base) * HH;  // guarded below
    if (t + 2 < T_STEPS)
      __builtin_prefetch(ins + ((size_t)(t + 2) * BB + base) * HH + tid * 16, 0, 1);
#pragma unroll
    for (int i = 0; i < 4; ++i) {
      int flat = (tid + i * THREADS) * 4;
      int m = flat >> 8, n = flat & 255;
      v4f g_r = *(const v4f*)(Gs + m * GSTRIDE + n);
      v4f g_z = *(const v4f*)(Gs + m * GSTRIDE + 256 + n);
      v4f g_i = *(const v4f*)(Gs + m * GSTRIDE + 512 + n);
      v4f g_h = *(const v4f*)(Gs + m * GSTRIDE + 768 + n);   // h@Whn + bhn
      v4h hold = *(const v4h*)(As + a_off(m, 256 + n));      // post-reset h
      v4f o;
#pragma unroll
      for (int q = 0; q < 4; ++q) {
        float r  = 1.0f / (1.0f + __expf(-g_r[q]));
        float z  = 1.0f / (1.0f + __expf(-g_z[q]));
        float nn = tanhf(g_i[q] + r * g_h[q]);
        o[q] = (1.0f - z) * nn + z * (float)hold[q];
      }
      *(v4f*)(outt + flat) = o;                              // y_t = h'
      if (t + 1 < T_STEPS) {                                 // stage step t+1
        int rst = resets[(t + 1) * BB + base + m];
        v4f xv = *(const v4f*)(xn + flat);
        v4h hp, xp;
#pragma unroll
        for (int q = 0; q < 4; ++q) {
          hp[q] = rst ? (f16)0.0f : (f16)o[q];
          xp[q] = (f16)xv[q];
        }
        *(v4h*)(As + a_off(m, 256 + n)) = hp;
        *(v4h*)(As + a_off(m, n))       = xp;
      }
    }
    __syncthreads();
  }
}

// ---------------------------------------------------------------------------
extern "C" void kernel_launch(void* const* d_in, const int* in_sizes, int n_in,
                              void* d_out, int out_size, void* d_ws, size_t ws_size,
                              hipStream_t stream) {
  const float* ins    = (const float*)d_in[0];
  const int*   resets = (const int*)  d_in[1];
  const float* carry  = (const float*)d_in[2];
  const float* Wi     = (const float*)d_in[3];
  const float* bi     = (const float*)d_in[4];
  const float* Wh     = (const float*)d_in[5];
  const float* bhn    = (const float*)d_in[6];

  f16* wtbuf = (f16*)d_ws;                    // 1 MB swizzled weights

  const int wt_elems = NJ * KCN * 32 * 16;    // 524288
  build_wt<<<wt_elems / 256, 256, 0, stream>>>(Wi, Wh, wtbuf);

  gru_scan<<<NWG, THREADS, LDS_BYTES, stream>>>(
      ins, resets, carry, wtbuf, bi, bhn, (float*)d_out);
}
